// EmotionGATv2_33904471835081
// MI455X (gfx1250) — matrix operand order repaired and checked by reading, not compiled, not run on hardware
//
#include <hip/hip_runtime.h>
#include <hip/hip_bf16.h>

// ---------------------------------------------------------------------------
// Problem constants (match reference)
// ---------------------------------------------------------------------------
#define NN      50000
#define EDGES   800000
#define F_IN    64
#define HEADS   4
#define C1      32
#define C2      64
#define D1      (HEADS * C1)   // 128
#define D2      (HEADS * C2)   // 256
#define NGRAPH  128
#define NCLS    8
#define EE      (EDGES + NN)   // edges + self loops

typedef __bf16 bf16_t;
typedef bf16_t v16bf __attribute__((ext_vector_type(16)));
typedef bf16_t v8bf  __attribute__((ext_vector_type(8)));
typedef float  v8f   __attribute__((ext_vector_type(8)));

// monotonic-uint encoding of float for atomicMax on unsigned
__device__ __forceinline__ unsigned f2mono(float f) {
  unsigned u = __float_as_uint(f);
  return (u & 0x80000000u) ? ~u : (u | 0x80000000u);
}
__device__ __forceinline__ float mono2f(unsigned u) {
  return (u & 0x80000000u) ? __uint_as_float(u & 0x7fffffffu)
                           : __uint_as_float(~u);
}
__device__ __forceinline__ unsigned short f2bf_rne(float f) {
  unsigned u = __float_as_uint(f);
  unsigned r = u + 0x7fffu + ((u >> 16) & 1u);
  return (unsigned short)(r >> 16);
}

// ---------------------------------------------------------------------------
// fp32 -> bf16 cast
// ---------------------------------------------------------------------------
__global__ void k_cast_bf16(const float* __restrict__ in,
                            unsigned short* __restrict__ out, long n) {
  long i = (long)blockIdx.x * blockDim.x + threadIdx.x;
  if (i < n) out[i] = f2bf_rne(in[i]);
}

// ---------------------------------------------------------------------------
// WMMA GEMM:  Out[M,OUT] = A[M,K](bf16) @ W[OUT,K]^T(bf16) + bias
// One wave32 per 16x16 output tile; K multiple of 32, M,OUT multiples of 16.
// A fragment (16x32 bf16, ISA 7.12.2): lanes 0-15 row M=l hold K 0-7 & 16-23,
// lanes 16-31 hold K 8-15 & 24-31.  B fragment (32x16): lane l holds column
// o=l%16, same K order as A (K 0-7 & 16-23 / 8-15 & 24-31) from W row o.
// ---------------------------------------------------------------------------
__global__ void k_wmma_gemm_bias(const bf16_t* __restrict__ A,
                                 const bf16_t* __restrict__ W,
                                 const float* __restrict__ bias,
                                 float* __restrict__ Out,
                                 int M, int K, int OUT) {
  int wave = (int)((blockIdx.x * blockDim.x + threadIdx.x) >> 5);
  int lane = threadIdx.x & 31;
  int tilesO = OUT >> 4;
  int tiles  = (M >> 4) * tilesO;
  if (wave >= tiles) return;               // wave-uniform: EXEC all-1 inside

  int tm = (wave / tilesO) << 4;
  int to = (wave % tilesO) << 4;
  int hi   = lane >> 4;                    // 0: lanes 0-15, 1: lanes 16-31
  int lmod = lane & 15;

  const bf16_t* arow = A + (long)(tm + lmod) * K;
  const bf16_t* brow = W + (long)(to + lmod) * K;

  v8f acc = {};
  for (int k0 = 0; k0 < K; k0 += 32) {
    int aoff = hi ? 8 : 0;
    v8bf a0 = *(const v8bf*)(arow + k0 + aoff);        // K aoff..aoff+7
    v8bf a1 = *(const v8bf*)(arow + k0 + aoff + 16);   // K aoff+16..aoff+23
    v16bf a;
#pragma unroll
    for (int i = 0; i < 8; ++i) { a[i] = a0[i]; a[8 + i] = a1[i]; }
    v8bf b0 = *(const v8bf*)(brow + k0 + aoff);
    v8bf b1 = *(const v8bf*)(brow + k0 + aoff + 16);
    v16bf b;
#pragma unroll
    for (int i = 0; i < 8; ++i) { b[i] = b0[i]; b[8 + i] = b1[i]; }
    acc = __builtin_amdgcn_wmma_f32_16x16x32_bf16(
        false, a, false, b, (short)0, acc, false, false);
  }

  int ocol = to + lmod;
  float bv = bias[ocol];
  float* orow = Out + (long)(tm + (hi ? 8 : 0)) * OUT + ocol;
#pragma unroll
  for (int r = 0; r < 8; ++r) orow[(long)r * OUT] = acc[r] + bv;
}

// ---------------------------------------------------------------------------
// Edge scoring: one wave per edge; lane owns CPL contiguous channels.
// e[edge,h] = sum_c att[h,c] * leakyrelu(xl[src,h,c] + xr[dst,h,c])
// Also atomic-max into per-(dst,head) running max (monotonic-uint encoding).
// Group of 8 lanes per head in both layers (CPL=4,C=32 / CPL=8,C=64).
// ---------------------------------------------------------------------------
template <int CPL>
__global__ void k_edge_score(const float* __restrict__ xl,
                             const float* __restrict__ xr,
                             const float* __restrict__ att,
                             const int* __restrict__ ei,
                             int Cc, float* __restrict__ eout,
                             unsigned* __restrict__ emax) {
  long wave = (long)(blockIdx.x * blockDim.x + threadIdx.x) >> 5;
  int lane = threadIdx.x & 31;
  if (wave >= EE) return;
  int s, d;
  if (wave < EDGES) { s = ei[wave]; d = ei[EDGES + wave]; }
  else              { s = d = (int)(wave - EDGES); }

  const int D = CPL * 32;
  int ch0 = lane * CPL;
  int head = ch0 / Cc;
  const float* xs = xl + (long)s * D + ch0;
  const float* xd = xr + (long)d * D + ch0;
  const float* at = att + ch0;           // att flat [H*C] == channel index

  float p = 0.f;
#pragma unroll
  for (int i = 0; i < CPL; ++i) {
    float v = xs[i] + xd[i];
    v = (v > 0.f) ? v : v * 0.2f;
    p += v * at[i];
  }
  // reduce across the 8 lanes of this head
  for (int off = 4; off > 0; off >>= 1) p += __shfl_xor(p, off, 32);
  if ((lane & 7) == 0) {
    eout[wave * HEADS + head] = p;
    atomicMax(&emax[d * HEADS + head], f2mono(p));
  }
}

// ---------------------------------------------------------------------------
// ex = exp(e - max[dst]); accumulate softmax denominator per (dst, head)
// ---------------------------------------------------------------------------
__global__ void k_edge_exp(const int* __restrict__ ei,
                           float* __restrict__ ebuf,
                           const unsigned* __restrict__ emax,
                           float* __restrict__ denom) {
  long t = (long)blockIdx.x * blockDim.x + threadIdx.x;
  if (t >= (long)EE * HEADS) return;
  long edge = t / HEADS;
  int  h    = (int)(t % HEADS);
  int d = (edge < EDGES) ? ei[EDGES + edge] : (int)(edge - EDGES);
  float ex = __expf(ebuf[t] - mono2f(emax[d * HEADS + h]));
  ebuf[t] = ex;
  atomicAdd(&denom[d * HEADS + h], ex);
}

// ---------------------------------------------------------------------------
// Scatter-accumulate: acc[dst] += (ex/denom[dst]) * xl[src]
// ---------------------------------------------------------------------------
template <int CPL>
__global__ void k_edge_aggregate(const float* __restrict__ xl,
                                 const int* __restrict__ ei, int Cc,
                                 const float* __restrict__ ebuf,
                                 const float* __restrict__ denom,
                                 float* __restrict__ acc) {
  long wave = (long)(blockIdx.x * blockDim.x + threadIdx.x) >> 5;
  int lane = threadIdx.x & 31;
  if (wave >= EE) return;
  int s, d;
  if (wave < EDGES) { s = ei[wave]; d = ei[EDGES + wave]; }
  else              { s = d = (int)(wave - EDGES); }

  const int D = CPL * 32;
  int ch0 = lane * CPL;
  int head = ch0 / Cc;
  float alpha = ebuf[wave * HEADS + head] / denom[d * HEADS + head];
  const float* xs = xl + (long)s * D + ch0;
  float* o = acc + (long)d * D + ch0;
#pragma unroll
  for (int i = 0; i < CPL; ++i) atomicAdd(&o[i], alpha * xs[i]);
}

// ---------------------------------------------------------------------------
// Layer-1 epilogue: bias + ELU, emit bf16 features for the next layer's GEMM
// ---------------------------------------------------------------------------
__global__ void k_finish_elu_cast(float* __restrict__ acc,
                                  const float* __restrict__ bias, int D,
                                  unsigned short* __restrict__ xbf) {
  long t = (long)blockIdx.x * blockDim.x + threadIdx.x;
  if (t >= (long)NN * D) return;
  int ch = (int)(t % D);
  float v = acc[t] + bias[ch];
  v = (v > 0.f) ? v : (__expf(v) - 1.f);
  acc[t] = v;
  xbf[t] = f2bf_rne(v);
}

// ---------------------------------------------------------------------------
// Layer-2 epilogue: bias + ELU, scatter into per-graph mean-pool accumulators
// ---------------------------------------------------------------------------
__global__ void k_finish_elu_pool(const float* __restrict__ acc,
                                  const float* __restrict__ bias,
                                  const int* __restrict__ batch, int D,
                                  float* __restrict__ sums,
                                  float* __restrict__ cnts) {
  long t = (long)blockIdx.x * blockDim.x + threadIdx.x;
  if (t >= (long)NN * D) return;
  int n  = (int)(t / D);
  int ch = (int)(t % D);
  float v = acc[t] + bias[ch];
  v = (v > 0.f) ? v : (__expf(v) - 1.f);
  int g = batch[n];
  atomicAdd(&sums[(long)g * D + ch], v);
  if (ch == 0) atomicAdd(&cnts[g], 1.0f);
}

// ---------------------------------------------------------------------------
// Classifier (tiny: 8 MFLOP total, plain fp32)
// ---------------------------------------------------------------------------
__global__ void k_fc1(const float* __restrict__ sums,
                      const float* __restrict__ cnts,
                      const float* __restrict__ W,
                      const float* __restrict__ b,
                      float* __restrict__ z) {
  int t = blockIdx.x * blockDim.x + threadIdx.x;
  if (t >= NGRAPH * 128) return;
  int g = t >> 7, j = t & 127;
  float inv = 1.0f / fmaxf(cnts[g], 1.0f);
  const float* p = sums + (long)g * D2;
  const float* w = W + (long)j * D2;
  float s = b[j];
  for (int k = 0; k < D2; ++k) s += (p[k] * inv) * w[k];
  z[t] = fmaxf(s, 0.f);
}

__global__ void k_fc2(const float* __restrict__ z,
                      const float* __restrict__ W,
                      const float* __restrict__ b,
                      float* __restrict__ out) {
  int t = blockIdx.x * blockDim.x + threadIdx.x;
  if (t >= NGRAPH * NCLS) return;
  int g = t >> 3, c = t & 7;
  const float* zp = z + (long)g * 128;
  const float* w = W + (long)c * 128;
  float s = b[c];
  for (int j = 0; j < 128; ++j) s += zp[j] * w[j];
  out[t] = s;
}

// ---------------------------------------------------------------------------
// Host orchestration
// ---------------------------------------------------------------------------
static inline size_t alignup(size_t v) { return (v + 255) & ~(size_t)255; }

extern "C" void kernel_launch(void* const* d_in, const int* in_sizes, int n_in,
                              void* d_out, int out_size, void* d_ws, size_t ws_size,
                              hipStream_t stream) {
  const float* x     = (const float*)d_in[0];
  const int*   ei    = (const int*)d_in[1];
  const int*   batch = (const int*)d_in[2];
  const float* Wl1 = (const float*)d_in[3];  const float* bl1 = (const float*)d_in[4];
  const float* Wr1 = (const float*)d_in[5];  const float* br1 = (const float*)d_in[6];
  const float* att1 = (const float*)d_in[7]; const float* bias1 = (const float*)d_in[8];
  const float* Wl2 = (const float*)d_in[9];  const float* bl2 = (const float*)d_in[10];
  const float* Wr2 = (const float*)d_in[11]; const float* br2 = (const float*)d_in[12];
  const float* att2 = (const float*)d_in[13]; const float* bias2 = (const float*)d_in[14];
  const float* Wf1 = (const float*)d_in[15]; const float* bf1v = (const float*)d_in[16];
  const float* Wf2 = (const float*)d_in[17]; const float* bf2v = (const float*)d_in[18];

  char* ws = (char*)d_ws;
  size_t o = 0;
  auto take = [&](size_t bytes) { size_t r = o; o += alignup(bytes); return r; };

  size_t oXBF  = take((size_t)NN * D1 * 2);        // bf16 layer input (layer2 reuse)
  size_t oWBl1 = take((size_t)D1 * F_IN * 2);
  size_t oWBr1 = take((size_t)D1 * F_IN * 2);
  size_t oWBl2 = take((size_t)D2 * D1 * 2);
  size_t oWBr2 = take((size_t)D2 * D1 * 2);
  size_t oXL   = take((size_t)NN * D2 * 4);
  size_t oXR   = take((size_t)NN * D2 * 4);
  size_t oACC  = take((size_t)NN * D2 * 4);
  size_t oEB   = take((size_t)EE * HEADS * 4);
  size_t oEMAX = take((size_t)NN * HEADS * 4);
  size_t oDEN  = take((size_t)NN * HEADS * 4);
  size_t oSUM  = take((size_t)NGRAPH * D2 * 4);
  size_t oCNT  = take((size_t)NGRAPH * 4);
  size_t oZ    = take((size_t)NGRAPH * 128 * 4);
  (void)ws_size; (void)in_sizes; (void)n_in; (void)out_size;

  unsigned short* XBF = (unsigned short*)(ws + oXBF);
  float* XL  = (float*)(ws + oXL);
  float* XR  = (float*)(ws + oXR);
  float* ACC = (float*)(ws + oACC);
  float* EB  = (float*)(ws + oEB);
  unsigned* EMAX = (unsigned*)(ws + oEMAX);
  float* DEN = (float*)(ws + oDEN);
  float* SUM = (float*)(ws + oSUM);
  float* CNT = (float*)(ws + oCNT);
  float* Z   = (float*)(ws + oZ);

  const int TB = 256;
  auto blk = [](long n, int tb) { return (unsigned)((n + tb - 1) / tb); };

  // ---- casts to bf16 ----
  k_cast_bf16<<<blk((long)NN * F_IN, TB), TB, 0, stream>>>(x, XBF, (long)NN * F_IN);
  k_cast_bf16<<<blk(D1 * F_IN, TB), TB, 0, stream>>>(Wl1, (unsigned short*)(ws + oWBl1), D1 * F_IN);
  k_cast_bf16<<<blk(D1 * F_IN, TB), TB, 0, stream>>>(Wr1, (unsigned short*)(ws + oWBr1), D1 * F_IN);
  k_cast_bf16<<<blk(D2 * D1, TB), TB, 0, stream>>>(Wl2, (unsigned short*)(ws + oWBl2), D2 * D1);
  k_cast_bf16<<<blk(D2 * D1, TB), TB, 0, stream>>>(Wr2, (unsigned short*)(ws + oWBr2), D2 * D1);

  // ================= Layer 1 =================
  {
    long tiles = ((long)NN / 16) * (D1 / 16);
    unsigned g = blk(tiles * 32, 128);
    k_wmma_gemm_bias<<<g, 128, 0, stream>>>((const bf16_t*)XBF, (const bf16_t*)(ws + oWBl1),
                                            bl1, XL, NN, F_IN, D1);
    k_wmma_gemm_bias<<<g, 128, 0, stream>>>((const bf16_t*)XBF, (const bf16_t*)(ws + oWBr1),
                                            br1, XR, NN, F_IN, D1);
  }
  hipMemsetAsync(EMAX, 0, (size_t)NN * HEADS * 4, stream);   // mono(0) < every float
  hipMemsetAsync(DEN, 0, (size_t)NN * HEADS * 4, stream);
  hipMemsetAsync(ACC, 0, (size_t)NN * D1 * 4, stream);

  {
    unsigned g = blk((long)EE * 32, 128);
    k_edge_score<4><<<g, 128, 0, stream>>>(XL, XR, att1, ei, C1, EB, EMAX);
    k_edge_exp<<<blk((long)EE * HEADS, TB), TB, 0, stream>>>(ei, EB, EMAX, DEN);
    k_edge_aggregate<4><<<g, 128, 0, stream>>>(XL, ei, C1, EB, DEN, ACC);
  }
  k_finish_elu_cast<<<blk((long)NN * D1, TB), TB, 0, stream>>>(ACC, bias1, D1, XBF);

  // ================= Layer 2 =================
  {
    long tiles = ((long)NN / 16) * (D2 / 16);
    unsigned g = blk(tiles * 32, 128);
    k_wmma_gemm_bias<<<g, 128, 0, stream>>>((const bf16_t*)XBF, (const bf16_t*)(ws + oWBl2),
                                            bl2, XL, NN, D1, D2);
    k_wmma_gemm_bias<<<g, 128, 0, stream>>>((const bf16_t*)XBF, (const bf16_t*)(ws + oWBr2),
                                            br2, XR, NN, D1, D2);
  }
  hipMemsetAsync(EMAX, 0, (size_t)NN * HEADS * 4, stream);
  hipMemsetAsync(DEN, 0, (size_t)NN * HEADS * 4, stream);
  hipMemsetAsync(ACC, 0, (size_t)NN * D2 * 4, stream);
  hipMemsetAsync(SUM, 0, (size_t)NGRAPH * D2 * 4, stream);
  hipMemsetAsync(CNT, 0, (size_t)NGRAPH * 4, stream);

  {
    unsigned g = blk((long)EE * 32, 128);
    k_edge_score<8><<<g, 128, 0, stream>>>(XL, XR, att2, ei, C2, EB, EMAX);
    k_edge_exp<<<blk((long)EE * HEADS, TB), TB, 0, stream>>>(ei, EB, EMAX, DEN);
    k_edge_aggregate<8><<<g, 128, 0, stream>>>(XL, ei, C2, EB, DEN, ACC);
  }
  k_finish_elu_pool<<<blk((long)NN * D2, TB), TB, 0, stream>>>(ACC, bias2, batch, D2, SUM, CNT);

  // ================= Classifier =================
  k_fc1<<<blk(NGRAPH * 128, TB), TB, 0, stream>>>(SUM, CNT, Wf1, bf1v, Z);
  k_fc2<<<blk(NGRAPH * NCLS, TB), TB, 0, stream>>>(Z, Wf2, bf2v, (float*)d_out);
}
